// MultiGranularityConsistencyModule_20830591385858
// MI455X (gfx1250) — matrix operand order repaired and checked by reading, not compiled, not run on hardware
//
#include <hip/hip_runtime.h>
#include <hip/hip_bf16.h>
#include <math.h>

// ---------------- problem constants ----------------
static constexpr int kB  = 4;
static constexpr int kC  = 640;
static constexpr int kH  = 48;
static constexpr int kW  = 48;
static constexpr int kHW = kH * kW;        // 2304
static constexpr int kK  = 40;             // top-k / fine-grained channels
static constexpr int kIC = 256;            // ASPP out channels
static constexpr int kC5 = 5 * kIC;        // 1280 concat channels
static constexpr int kOC = 2 * kIC + 2 * kK; // 592 output channels

typedef __attribute__((ext_vector_type(16))) __bf16 v16bf;
typedef __attribute__((ext_vector_type(8)))  float  v8f;

__device__ __forceinline__ __bf16 f2bf(float f) {
  union { float f; unsigned u; } v; v.f = f;
  unsigned r = v.u + 0x7FFFu + ((v.u >> 16) & 1u);   // round-to-nearest-even
  unsigned short h = (unsigned short)(r >> 16);
  union { unsigned short s; __bf16 b; } o; o.s = h;
  return o.b;
}

// ---------------- WMMA fragment loads from LDS ------------------
// A tile (16x32 bf16, row-major, row stride ld):
// lane<16 holds row=lane, K in {0..15}; lane>=16 holds row=lane-16,
// K in {16..31}.
__device__ __forceinline__ v16bf load_frag_A(const __bf16* p, int ld, int lane) {
  int r  = lane & 15;
  int kh = (lane >> 4) * 16;
  v16bf a;
#pragma unroll
  for (int e = 0; e < 16; ++e) {
    int k = kh + e;
    a[e] = p[r * ld + k];
  }
  return a;
}
// B tile (32x16 bf16, row-major rows=k, row stride ld): lanes 0-15 hold
// K=0..15, lanes 16-31 hold K=16..31, col = lane&15.
__device__ __forceinline__ v16bf load_frag_B(const __bf16* p, int ld, int lane) {
  int c  = lane & 15;
  int kb = (lane >> 4) * 16;
  v16bf b;
#pragma unroll
  for (int e = 0; e < 16; ++e) b[e] = p[(kb + e) * ld + c];
  return b;
}

// ---------------- 1: normalize + bf16 staging ----------------
__global__ __launch_bounds__(256) void prep_norm(const float* __restrict__ x,
                                                 __bf16* __restrict__ xb,
                                                 __bf16* __restrict__ fncn,
                                                 __bf16* __restrict__ fnnc) {
  int g = blockIdx.x * 256 + threadIdx.x;      // b*HW + n
  int b = g / kHW, n = g % kHW;
  const float* xp = x + (size_t)b * kC * kHW + n;
  float ss = 0.f;
  for (int c = 0; c < kC; ++c) { float v = xp[(size_t)c * kHW]; ss += v * v; }
  float inv = 1.0f / fmaxf(sqrtf(ss), 1e-12f);
  for (int c = 0; c < kC; ++c) {
    float v = xp[(size_t)c * kHW];
    size_t icn = (size_t)(b * kC + c) * kHW + n;
    xb[icn] = f2bf(v);
    __bf16 fb = f2bf(v * inv);
    fncn[icn] = fb;
    fnnc[(size_t)(b * kHW + n) * kC + c] = fb;
  }
}

// ---------------- weight conversions ----------------
__global__ void cvt_bf16(const float* __restrict__ s, __bf16* __restrict__ d, int n) {
  int i = blockIdx.x * 256 + threadIdx.x;
  if (i < n) d[i] = f2bf(s[i]);
}
// (oc,ci,kh,kw) fp32  ->  [tap][oc][ci] bf16
__global__ void cvt_tap(const float* __restrict__ s, __bf16* __restrict__ d) {
  int i = blockIdx.x * 256 + threadIdx.x;
  if (i < 9 * kIC * kC) {
    int t = i / (kIC * kC);
    int rem = i % (kIC * kC);
    int oc = rem / kC, ci = rem % kC;
    d[i] = f2bf(s[(oc * kC + ci) * 9 + t]);
  }
}

// ---------------- 2: correlation GEMM (bf16 WMMA) ----------------
// Abuf[b][n][m] = alpha * mask(n,m) * sum_c fn[b][c][n] * fn[b][c][m]
__global__ __launch_bounds__(256) void corr_gemm(const __bf16* __restrict__ fnnc,
                                                 const __bf16* __restrict__ fncn,
                                                 float* __restrict__ Abuf,
                                                 const float* __restrict__ alphap) {
  __shared__ __bf16 As[128 * 32];
  __shared__ __bf16 Bs[32 * 128];
  int b  = blockIdx.z;
  int n0 = blockIdx.x * 128;   // output column m
  int r0 = blockIdx.y * 128;   // output row n (query)
  int tid = threadIdx.x, lane = tid & 31, wave = tid >> 5;
  int wr = wave >> 1, wc = wave & 1;
  v8f acc[2][4];
#pragma unroll
  for (int i = 0; i < 2; ++i)
#pragma unroll
    for (int j = 0; j < 4; ++j)
#pragma unroll
      for (int e = 0; e < 8; ++e) acc[i][j][e] = 0.f;

  const __bf16* Ab = fnnc + (size_t)b * kHW * kC;
  const __bf16* Xb = fncn + (size_t)b * kC * kHW;
  for (int kc = 0; kc < kC; kc += 32) {
#pragma unroll
    for (int i = 0; i < 2; ++i) {
      int chunk = tid + i * 256;
      int r = chunk >> 2, kq = (chunk & 3) * 8;
      const __bf16* src = Ab + (size_t)(r0 + r) * kC + kc + kq;
      *(uint4*)&As[r * 32 + kq] = *(const uint4*)src;
    }
#pragma unroll
    for (int i = 0; i < 2; ++i) {
      int chunk = tid + i * 256;
      int kk = chunk >> 4, nq = (chunk & 15) * 8;
      const __bf16* src = Xb + (size_t)(kc + kk) * kHW + n0 + nq;
      *(uint4*)&Bs[kk * 128 + nq] = *(const uint4*)src;
    }
    if (kc + 32 < kC) __builtin_prefetch(Xb + (size_t)(kc + 32) * kHW + n0, 0, 1);
    __syncthreads();
    v16bf af[2], bfr[4];
#pragma unroll
    for (int i = 0; i < 2; ++i) af[i]  = load_frag_A(&As[(wr * 32 + i * 16) * 32], 32, lane);
#pragma unroll
    for (int j = 0; j < 4; ++j) bfr[j] = load_frag_B(&Bs[wc * 64 + j * 16], 128, lane);
#pragma unroll
    for (int i = 0; i < 2; ++i)
#pragma unroll
      for (int j = 0; j < 4; ++j)
        acc[i][j] = __builtin_amdgcn_wmma_f32_16x16x32_bf16(
            false, af[i], false, bfr[j], (short)0, acc[i][j], false, false);
    __syncthreads();
  }
  float alpha = *alphap;
  const float inv2s2 = 1.f / (2.f * 2.4f * 2.4f);  // sr = sc = 48*0.05
  int colL = lane & 15, rowO = (lane >> 4) * 8;
#pragma unroll
  for (int i = 0; i < 2; ++i)
#pragma unroll
    for (int j = 0; j < 4; ++j)
#pragma unroll
      for (int v = 0; v < 8; ++v) {
        int n = r0 + wr * 32 + i * 16 + rowO + v;
        int m = n0 + wc * 64 + j * 16 + colL;
        float di = (float)(m / kW) - (float)(n / kW);
        float dj = (float)(m % kW) - (float)(n % kW);
        float msk = 1.f - __expf(-(di * di) * inv2s2) * __expf(-(dj * dj) * inv2s2);
        Abuf[((size_t)b * kHW + n) * kHW + m] = alpha * msk * acc[i][j][v];
      }
}

// ---------------- softmax statistics ----------------
__global__ __launch_bounds__(256) void row_stats(const float* __restrict__ Abuf,
                                                 float* __restrict__ rmax,
                                                 float* __restrict__ rsum) {
  int bn = blockIdx.x;
  const float* row = Abuf + (size_t)bn * kHW;
  float M = -3.4e38f, S = 0.f;
  for (int m = threadIdx.x; m < kHW; m += 256) {
    float v = row[m];
    float Mn = fmaxf(M, v);
    S = S * __expf(M - Mn) + __expf(v - Mn);
    M = Mn;
  }
  __shared__ float sM[256], sS[256];
  sM[threadIdx.x] = M; sS[threadIdx.x] = S;
  __syncthreads();
  for (int s = 128; s > 0; s >>= 1) {
    if (threadIdx.x < (unsigned)s) {
      float m2 = sM[threadIdx.x + s], s2 = sS[threadIdx.x + s];
      float M1 = sM[threadIdx.x], S1 = sS[threadIdx.x];
      float Mn = fmaxf(M1, m2);
      sS[threadIdx.x] = S1 * __expf(M1 - Mn) + s2 * __expf(m2 - Mn);
      sM[threadIdx.x] = Mn;
    }
    __syncthreads();
  }
  if (threadIdx.x == 0) { rmax[bn] = sM[0]; rsum[bn] = sS[0]; }
}

__global__ __launch_bounds__(256) void col_stats(const float* __restrict__ Abuf,
                                                 float* __restrict__ cmax,
                                                 float* __restrict__ csum) {
  int b = blockIdx.y;
  int m = blockIdx.x * 256 + threadIdx.x;
  const float* base = Abuf + (size_t)b * kHW * kHW + m;
  float M = -3.4e38f, S = 0.f;
  for (int n = 0; n < kHW; ++n) {
    float v = base[(size_t)n * kHW];
    float Mn = fmaxf(M, v);
    S = S * __expf(M - Mn) + __expf(v - Mn);
    M = Mn;
  }
  cmax[b * kHW + m] = M; csum[b * kHW + m] = S;
}

__global__ __launch_bounds__(256) void ga_kernel(float* __restrict__ Abuf,
                                                 const float* __restrict__ rmax,
                                                 const float* __restrict__ rsum,
                                                 const float* __restrict__ cmax,
                                                 const float* __restrict__ csum) {
  size_t total = (size_t)kB * kHW * kHW;
  for (size_t idx = (size_t)blockIdx.x * 256 + threadIdx.x; idx < total;
       idx += (size_t)gridDim.x * 256) {
    size_t bn = idx / kHW; int m = (int)(idx % kHW);
    int n = (int)(bn % kHW); int b = (int)(bn / kHW);
    float a = Abuf[idx];
    float g = __expf(a - rmax[b * kHW + n]) * __expf(a - cmax[b * kHW + m]) /
              (rsum[b * kHW + n] * csum[b * kHW + m]);
    Abuf[idx] = g;
  }
}

__global__ __launch_bounds__(256) void colsum_kernel(const float* __restrict__ ga,
                                                     float* __restrict__ csum2) {
  int b = blockIdx.y;
  int m = blockIdx.x * 256 + threadIdx.x;
  const float* base = ga + (size_t)b * kHW * kHW + m;
  float s = 0.f;
  for (int n = 0; n < kHW; ++n) s += base[(size_t)n * kHW];
  csum2[b * kHW + m] = s;
}

// ---------------- top-k (K=40, descending) ----------------
__global__ __launch_bounds__(256) void topk_kernel(const float* __restrict__ ga,
                                                   float* __restrict__ f1,
                                                   float* __restrict__ dout) {
  int b = blockIdx.y;
  int m = blockIdx.x * 256 + threadIdx.x;
  const float* base = ga + (size_t)b * kHW * kHW + m;
  float tv[kK];
  for (int k = 0; k < kK; ++k) tv[k] = -3.4e38f;
  for (int n = 0; n < kHW; ++n) {
    float v = base[(size_t)n * kHW];
    if (v > tv[kK - 1]) {
      int k = kK - 1;
      while (k > 0 && tv[k - 1] < v) { tv[k] = tv[k - 1]; --k; }
      tv[k] = v;
    }
  }
  for (int k = 0; k < kK; ++k) {
    float v = tv[k];
    f1[((size_t)b * kK + k) * kHW + m] = v;
    dout[((size_t)b * kOC + 2 * kIC + k) * kHW + m] = v;
  }
}

// ---------------- FineGrained module ----------------
__global__ __launch_bounds__(256) void fg_t_kernel(const float* __restrict__ f1,
                                                   const float* __restrict__ w,
                                                   const float* __restrict__ s,
                                                   const float* __restrict__ bo,
                                                   float* __restrict__ tb) {
  int b = blockIdx.y;
  int m = blockIdx.x * 256 + threadIdx.x;
  float fr[kK];
  for (int k = 0; k < kK; ++k) fr[k] = f1[((size_t)b * kK + k) * kHW + m];
  for (int ko = 0; ko < kK; ++ko) {
    float acc = 0.f;
    for (int ki = 0; ki < kK; ++ki) acc += w[ko * kK + ki] * fr[ki];
    tb[((size_t)b * kK + ko) * kHW + m] = fmaxf(s[ko] * acc + bo[ko], 0.f);
  }
}

__global__ __launch_bounds__(256) void tmean_kernel(const float* __restrict__ tb,
                                                    float* __restrict__ tm) {
  int b = blockIdx.y, k = blockIdx.x;
  const float* p = tb + ((size_t)b * kK + k) * kHW;
  float s = 0.f;
  for (int m = threadIdx.x; m < kHW; m += 256) s += p[m];
  __shared__ float sh[256];
  sh[threadIdx.x] = s; __syncthreads();
  for (int st = 128; st > 0; st >>= 1) {
    if (threadIdx.x < (unsigned)st) sh[threadIdx.x] += sh[threadIdx.x + st];
    __syncthreads();
  }
  if (threadIdx.x == 0) tm[b * kK + k] = sh[0] / (float)kHW;
}

__global__ void eca_kernel(const float* __restrict__ tm, const float* __restrict__ w,
                           float* __restrict__ y) {
  int t = threadIdx.x;
  if (t >= kB * kK) return;
  int b = t / kK, k = t % kK;
  float a = 0.f;
  for (int d = 0; d < 3; ++d) {
    int kk = k + d - 1;
    if (kk >= 0 && kk < kK) a += w[d] * tm[b * kK + kk];
  }
  y[t] = 1.f / (1.f + __expf(-a));
}

__global__ __launch_bounds__(256) void fg_f2_kernel(const float* __restrict__ tb,
                                                    const float* __restrict__ y,
                                                    const float* __restrict__ w,
                                                    const float* __restrict__ s,
                                                    const float* __restrict__ bo,
                                                    float* __restrict__ f2,
                                                    float* __restrict__ dout) {
  int b = blockIdx.y;
  int m = blockIdx.x * 256 + threadIdx.x;
  float fr[kK];
  for (int k = 0; k < kK; ++k)
    fr[k] = tb[((size_t)b * kK + k) * kHW + m] * y[b * kK + k];
  for (int ko = 0; ko < kK; ++ko) {
    float acc = 0.f;
    for (int ki = 0; ki < kK; ++ki) acc += w[ko * kK + ki] * fr[ki];
    float v = fmaxf(s[ko] * acc + bo[ko], 0.f);
    f2[((size_t)b * kK + ko) * kHW + m] = v;
    dout[((size_t)b * kOC + 2 * kIC + kK + ko) * kHW + m] = v;
  }
}

// ---------------- conv_select ----------------
__global__ __launch_bounds__(256) void cs12_kernel(const float* __restrict__ in, int cin,
                                                   const float* __restrict__ w,
                                                   const float* __restrict__ cb,
                                                   const float* __restrict__ s,
                                                   const float* __restrict__ o,
                                                   float* __restrict__ out) {
  int b = blockIdx.y;
  int m = blockIdx.x * 256 + threadIdx.x;
  int i = m / kW, j = m % kW;
  float acc[16];
#pragma unroll
  for (int c = 0; c < 16; ++c) acc[c] = 0.f;
  for (int ci = 0; ci < cin; ++ci) {
    const float* xp = in + ((size_t)b * cin + ci) * kHW;
    for (int kh = 0; kh < 3; ++kh) {
      int ii = i + kh - 1;
      if ((unsigned)ii >= (unsigned)kH) continue;
      for (int kw = 0; kw < 3; ++kw) {
        int jj = j + kw - 1;
        if ((unsigned)jj >= (unsigned)kW) continue;
        float xv = xp[ii * kW + jj];
#pragma unroll
        for (int oc = 0; oc < 16; ++oc)
          acc[oc] += w[((oc * cin + ci) * 3 + kh) * 3 + kw] * xv;
      }
    }
  }
#pragma unroll
  for (int oc = 0; oc < 16; ++oc)
    out[((size_t)b * 16 + oc) * kHW + m] = fmaxf(s[oc] * (acc[oc] + cb[oc]) + o[oc], 0.f);
}

__global__ __launch_bounds__(256) void cs34_kernel(const float* __restrict__ s2,
                                                   const float* __restrict__ w3,
                                                   const float* __restrict__ b3,
                                                   const float* __restrict__ w4,
                                                   const float* __restrict__ b4,
                                                   float* __restrict__ la) {
  int b = blockIdx.y;
  int m = blockIdx.x * 256 + threadIdx.x;
  int i = m / kW, j = m % kW;
  float acc[16];
#pragma unroll
  for (int c = 0; c < 16; ++c) acc[c] = 0.f;
  for (int ci = 0; ci < 16; ++ci) {
    const float* xp = s2 + ((size_t)b * 16 + ci) * kHW;
    for (int kh = 0; kh < 3; ++kh) {
      int ii = i + kh - 1;
      if ((unsigned)ii >= (unsigned)kH) continue;
      for (int kw = 0; kw < 3; ++kw) {
        int jj = j + kw - 1;
        if ((unsigned)jj >= (unsigned)kW) continue;
        float xv = xp[ii * kW + jj];
#pragma unroll
        for (int oc = 0; oc < 16; ++oc)
          acc[oc] += w3[((oc * 16 + ci) * 3 + kh) * 3 + kw] * xv;
      }
    }
  }
  float r = b4[0];
#pragma unroll
  for (int oc = 0; oc < 16; ++oc) r += w4[oc] * (acc[oc] + b3[oc]);
  la[(size_t)b * kHW + m] = 1.f / (1.f + __expf(-r));
}

// ---------------- pooled ASPP branch ----------------
__global__ __launch_bounds__(256) void gp_kernel(const float* __restrict__ x,
                                                 float* __restrict__ gp) {
  int b = blockIdx.y, c = blockIdx.x;
  const float* p = x + ((size_t)b * kC + c) * kHW;
  float s = 0.f;
  for (int m = threadIdx.x; m < kHW; m += 256) s += p[m];
  __shared__ float sh[256];
  sh[threadIdx.x] = s; __syncthreads();
  for (int st = 128; st > 0; st >>= 1) {
    if (threadIdx.x < (unsigned)st) sh[threadIdx.x] += sh[threadIdx.x + st];
    __syncthreads();
  }
  if (threadIdx.x == 0) gp[b * kC + c] = sh[0] / (float)kHW;
}

__global__ void bpv_kernel(const float* __restrict__ gp, const float* __restrict__ w,
                           const float* __restrict__ s, const float* __restrict__ bo,
                           float* __restrict__ bpv) {
  int b = blockIdx.x, oc = threadIdx.x;  // 256 threads
  const float* g = gp + (size_t)b * kC;
  float acc = 0.f;
  for (int c = 0; c < kC; ++c) acc += w[oc * kC + c] * g[c];
  bpv[b * kIC + oc] = fmaxf(s[oc] * acc + bo[oc], 0.f);
}

__global__ __launch_bounds__(256) void bp_bcast(const float* __restrict__ bpv,
                                                __bf16* __restrict__ concat) {
  size_t total = (size_t)kB * kIC * kHW;
  for (size_t idx = (size_t)blockIdx.x * 256 + threadIdx.x; idx < total;
       idx += (size_t)gridDim.x * 256) {
    size_t bo = idx / kHW; int m = (int)(idx % kHW);
    int oc = (int)(bo % kIC); int b = (int)(bo / kIC);
    concat[((size_t)b * kC5 + 4 * kIC + oc) * kHW + m] = f2bf(bpv[b * kIC + oc]);
  }
}

// ---------------- generic WMMA implicit-GEMM ----------------
// out[b][oc][m] = sum_{tap, ci} A[tap][oc][ci] * X[b][ci][shift_tap(m)]
// mode 0: relu(scale*acc+bias)            -> bf16 concat at choff_bf
// mode 1: relu(scale*acc+bias)*la[b][m]   -> fp32 d_out (f3) + bf16 f3b
// mode 2: raw acc                         -> fp32 d_out (f4)
__global__ __launch_bounds__(256) void gemm_generic(
    const __bf16* __restrict__ A, size_t aBatch, size_t tapStride,
    const __bf16* __restrict__ X, int cin, int ntaps, int dil,
    const float* __restrict__ scale, const float* __restrict__ bias,
    __bf16* __restrict__ outbf, int choff_bf,
    float* __restrict__ outf, int choff_f,
    __bf16* __restrict__ f3bOut, const float* __restrict__ la, int mode) {
  __shared__ __bf16 As[128 * 32];
  __shared__ __bf16 Bs[32 * 128];
  int b  = blockIdx.z;
  int n0 = blockIdx.x * 128;
  int r0 = blockIdx.y * 128;
  int tid = threadIdx.x, lane = tid & 31, wave = tid >> 5;
  int wr = wave >> 1, wc = wave & 1;
  v8f acc[2][4];
#pragma unroll
  for (int i = 0; i < 2; ++i)
#pragma unroll
    for (int j = 0; j < 4; ++j)
#pragma unroll
      for (int e = 0; e < 8; ++e) acc[i][j][e] = 0.f;

  const __bf16* Ab = A + (size_t)b * aBatch;
  for (int t = 0; t < ntaps; ++t) {
    int dh = (ntaps > 1) ? (t / 3 - 1) : 0;
    int dw = (ntaps > 1) ? (t % 3 - 1) : 0;
    const __bf16* At = Ab + (size_t)t * tapStride;
    for (int kc = 0; kc < cin; kc += 32) {
#pragma unroll
      for (int i = 0; i < 2; ++i) {
        int chunk = tid + i * 256;
        int r = chunk >> 2, kq = (chunk & 3) * 8;
        const __bf16* src = At + (size_t)(r0 + r) * cin + kc + kq;
        *(uint4*)&As[r * 32 + kq] = *(const uint4*)src;
      }
      if (ntaps == 1) {
#pragma unroll
        for (int i = 0; i < 2; ++i) {
          int chunk = tid + i * 256;
          int kk = chunk >> 4, nq = (chunk & 15) * 8;
          const __bf16* src = X + ((size_t)b * cin + kc + kk) * kHW + n0 + nq;
          *(uint4*)&Bs[kk * 128 + nq] = *(const uint4*)src;
        }
        if (kc + 32 < cin)
          __builtin_prefetch(X + ((size_t)b * cin + kc + 32) * kHW + n0, 0, 1);
      } else {
        for (int e = tid; e < 4096; e += 256) {
          int kk = e >> 7, nn = e & 127;
          int m = n0 + nn;
          int pi = m / kW + dil * dh;
          int pj = m % kW + dil * dw;
          __bf16 v = f2bf(0.f);
          if ((unsigned)pi < (unsigned)kH && (unsigned)pj < (unsigned)kW)
            v = X[((size_t)b * cin + kc + kk) * kHW + pi * kW + pj];
          Bs[kk * 128 + nn] = v;
        }
      }
      __syncthreads();
      v16bf af[2], bfr[4];
#pragma unroll
      for (int i = 0; i < 2; ++i) af[i]  = load_frag_A(&As[(wr * 32 + i * 16) * 32], 32, lane);
#pragma unroll
      for (int j = 0; j < 4; ++j) bfr[j] = load_frag_B(&Bs[wc * 64 + j * 16], 128, lane);
#pragma unroll
      for (int i = 0; i < 2; ++i)
#pragma unroll
        for (int j = 0; j < 4; ++j)
          acc[i][j] = __builtin_amdgcn_wmma_f32_16x16x32_bf16(
              false, af[i], false, bfr[j], (short)0, acc[i][j], false, false);
      __syncthreads();
    }
  }
  int colL = lane & 15, rowO = (lane >> 4) * 8;
#pragma unroll
  for (int i = 0; i < 2; ++i)
#pragma unroll
    for (int j = 0; j < 4; ++j)
#pragma unroll
      for (int v = 0; v < 8; ++v) {
        int R  = r0 + wr * 32 + i * 16 + rowO + v;  // output channel
        int Cc = n0 + wc * 64 + j * 16 + colL;      // pixel
        float a = acc[i][j][v];
        if (mode == 0) {
          float val = fmaxf(scale[R] * a + bias[R], 0.f);
          outbf[((size_t)b * kC5 + choff_bf + R) * kHW + Cc] = f2bf(val);
        } else if (mode == 1) {
          float val = fmaxf(scale[R] * a + bias[R], 0.f) * la[(size_t)b * kHW + Cc];
          outf[((size_t)b * kOC + choff_f + R) * kHW + Cc] = val;
          f3bOut[((size_t)b * kIC + R) * kHW + Cc] = f2bf(val);
        } else {
          outf[((size_t)b * kOC + choff_f + R) * kHW + Cc] = a;
        }
      }
}

// ---------------- gan (column-normalized ga, bf16) ----------------
__global__ __launch_bounds__(256) void gan_kernel(const float* __restrict__ ga,
                                                  const float* __restrict__ csum2,
                                                  __bf16* __restrict__ ganb) {
  size_t total = (size_t)kB * kHW * kHW;
  for (size_t idx = (size_t)blockIdx.x * 256 + threadIdx.x; idx < total;
       idx += (size_t)gridDim.x * 256) {
    size_t bn = idx / kHW; int m = (int)(idx % kHW);
    int b = (int)(bn / kHW);
    ganb[idx] = f2bf(ga[idx] / (csum2[b * kHW + m] + 1e-8f));
  }
}

// ================= host launcher =================
extern "C" void kernel_launch(void* const* d_in, const int* in_sizes, int n_in,
                              void* d_out, int out_size, void* d_ws, size_t ws_size,
                              hipStream_t stream) {
  (void)in_sizes; (void)n_in; (void)out_size; (void)ws_size;
  // inputs in setup_inputs() dict order: x, then params values in insertion order
  const float* x        = (const float*)d_in[0];
  const float* alphap   = (const float*)d_in[1];
  const float* fg_pre_w = (const float*)d_in[2];
  const float* fg_pre_s = (const float*)d_in[3];
  const float* fg_pre_b = (const float*)d_in[4];
  const float* fg_eca_w = (const float*)d_in[5];
  const float* fg_nxt_w = (const float*)d_in[6];
  const float* fg_nxt_s = (const float*)d_in[7];
  const float* fg_nxt_b = (const float*)d_in[8];
  const float* cs1_w = (const float*)d_in[9];
  const float* cs1_b = (const float*)d_in[10];
  const float* cs1_s = (const float*)d_in[11];
  const float* cs1_o = (const float*)d_in[12];
  const float* cs2_w = (const float*)d_in[13];
  const float* cs2_b = (const float*)d_in[14];
  const float* cs2_s = (const float*)d_in[15];
  const float* cs2_o = (const float*)d_in[16];
  const float* cs3_w = (const float*)d_in[17];
  const float* cs3_b = (const float*)d_in[18];
  const float* cs4_w = (const float*)d_in[19];
  const float* cs4_b = (const float*)d_in[20];
  const float* a0_w = (const float*)d_in[21];
  const float* a0_s = (const float*)d_in[22];
  const float* a0_b = (const float*)d_in[23];
  const float* a1_w = (const float*)d_in[24];
  const float* a1_s = (const float*)d_in[25];
  const float* a1_b = (const float*)d_in[26];
  const float* a2_w = (const float*)d_in[27];
  const float* a2_s = (const float*)d_in[28];
  const float* a2_b = (const float*)d_in[29];
  const float* a3_w = (const float*)d_in[30];
  const float* a3_s = (const float*)d_in[31];
  const float* a3_b = (const float*)d_in[32];
  const float* ap_w = (const float*)d_in[33];
  const float* ap_s = (const float*)d_in[34];
  const float* ap_b = (const float*)d_in[35];
  const float* apr_w = (const float*)d_in[36];
  const float* apr_s = (const float*)d_in[37];
  const float* apr_b = (const float*)d_in[38];
  float* dout = (float*)d_out;

  char* w = (char*)d_ws;
  size_t off = 0;
  auto alloc = [&](size_t bytes) -> char* {
    char* p = w + off;
    off += (bytes + 255) & ~(size_t)255;
    return p;
  };
  __bf16* xb    = (__bf16*)alloc((size_t)kB * kC * kHW * 2);
  __bf16* fncn  = (__bf16*)alloc((size_t)kB * kC * kHW * 2);
  __bf16* fnnc  = (__bf16*)alloc((size_t)kB * kHW * kC * 2);
  float*  Abuf  = (float*)alloc((size_t)kB * kHW * kHW * 4);
  float*  rmax  = (float*)alloc((size_t)kB * kHW * 4);
  float*  rsum  = (float*)alloc((size_t)kB * kHW * 4);
  float*  cmax  = (float*)alloc((size_t)kB * kHW * 4);
  float*  csumx = (float*)alloc((size_t)kB * kHW * 4);
  float*  csum2 = (float*)alloc((size_t)kB * kHW * 4);
  float*  f1buf = (float*)alloc((size_t)kB * kK * kHW * 4);
  float*  tbuf  = (float*)alloc((size_t)kB * kK * kHW * 4);
  float*  f2buf = (float*)alloc((size_t)kB * kK * kHW * 4);
  float*  tmean = (float*)alloc((size_t)kB * kK * 4);
  float*  ybuf  = (float*)alloc((size_t)kB * kK * 4);
  float*  s1    = (float*)alloc((size_t)kB * 16 * kHW * 4);
  float*  s2    = (float*)alloc((size_t)kB * 16 * kHW * 4);
  float*  labuf = (float*)alloc((size_t)kB * kHW * 4);
  float*  gpbuf = (float*)alloc((size_t)kB * kC * 4);
  float*  bpv   = (float*)alloc((size_t)kB * kIC * 4);
  __bf16* wba0  = (__bf16*)alloc((size_t)kIC * kC * 2);
  __bf16* wba1  = (__bf16*)alloc((size_t)9 * kIC * kC * 2);
  __bf16* wba2  = (__bf16*)alloc((size_t)9 * kIC * kC * 2);
  __bf16* wba3  = (__bf16*)alloc((size_t)9 * kIC * kC * 2);
  __bf16* wbapr = (__bf16*)alloc((size_t)kIC * kC5 * 2);
  __bf16* concat= (__bf16*)alloc((size_t)kB * kC5 * kHW * 2);
  __bf16* f3b   = (__bf16*)alloc((size_t)kB * kIC * kHW * 2);
  __bf16* ganb  = (__bf16*)alloc((size_t)kB * kHW * kHW * 2);

  // ---- stage inputs / weights ----
  prep_norm<<<(kB * kHW) / 256, 256, 0, stream>>>(x, xb, fncn, fnnc);
  cvt_bf16<<<(kIC * kC + 255) / 256, 256, 0, stream>>>(a0_w, wba0, kIC * kC);
  cvt_tap<<<(9 * kIC * kC + 255) / 256, 256, 0, stream>>>(a1_w, wba1);
  cvt_tap<<<(9 * kIC * kC + 255) / 256, 256, 0, stream>>>(a2_w, wba2);
  cvt_tap<<<(9 * kIC * kC + 255) / 256, 256, 0, stream>>>(a3_w, wba3);
  cvt_bf16<<<(kIC * kC5 + 255) / 256, 256, 0, stream>>>(apr_w, wbapr, kIC * kC5);

  // ---- attention ----
  corr_gemm<<<dim3(kHW / 128, kHW / 128, kB), 256, 0, stream>>>(fnnc, fncn, Abuf, alphap);
  row_stats<<<kB * kHW, 256, 0, stream>>>(Abuf, rmax, rsum);
  col_stats<<<dim3(kHW / 256, kB), 256, 0, stream>>>(Abuf, cmax, csumx);
  ga_kernel<<<4096, 256, 0, stream>>>(Abuf, rmax, rsum, cmax, csumx);
  colsum_kernel<<<dim3(kHW / 256, kB), 256, 0, stream>>>(Abuf, csum2);
  topk_kernel<<<dim3(kHW / 256, kB), 256, 0, stream>>>(Abuf, f1buf, dout);

  // ---- fine-grained module ----
  fg_t_kernel<<<dim3(kHW / 256, kB), 256, 0, stream>>>(f1buf, fg_pre_w, fg_pre_s, fg_pre_b, tbuf);
  tmean_kernel<<<dim3(kK, kB), 256, 0, stream>>>(tbuf, tmean);
  eca_kernel<<<1, 256, 0, stream>>>(tmean, fg_eca_w, ybuf);
  fg_f2_kernel<<<dim3(kHW / 256, kB), 256, 0, stream>>>(tbuf, ybuf, fg_nxt_w, fg_nxt_s, fg_nxt_b, f2buf, dout);

  // ---- conv_select -> local attention ----
  cs12_kernel<<<dim3(kHW / 256, kB), 256, 0, stream>>>(f2buf, kK, cs1_w, cs1_b, cs1_s, cs1_o, s1);
  cs12_kernel<<<dim3(kHW / 256, kB), 256, 0, stream>>>(s1, 16, cs2_w, cs2_b, cs2_s, cs2_o, s2);
  cs34_kernel<<<dim3(kHW / 256, kB), 256, 0, stream>>>(s2, cs3_w, cs3_b, cs4_w, cs4_b, labuf);

  // ---- ASPP ----
  gp_kernel<<<dim3(kC, kB), 256, 0, stream>>>(x, gpbuf);
  bpv_kernel<<<kB, 256, 0, stream>>>(gpbuf, ap_w, ap_s, ap_b, bpv);
  bp_bcast<<<4096, 256, 0, stream>>>(bpv, concat);

  dim3 gconv(kHW / 128, kIC / 128, kB);
  gemm_generic<<<gconv, 256, 0, stream>>>(wba0, 0, 0, xb, kC, 1, 0, a0_s, a0_b,
                                          concat, 0, nullptr, 0, nullptr, nullptr, 0);
  gemm_generic<<<gconv, 256, 0, stream>>>(wba1, 0, (size_t)kIC * kC, xb, kC, 9, 12, a1_s, a1_b,
                                          concat, kIC, nullptr, 0, nullptr, nullptr, 0);
  gemm_generic<<<gconv, 256, 0, stream>>>(wba2, 0, (size_t)kIC * kC, xb, kC, 9, 24, a2_s, a2_b,
                                          concat, 2 * kIC, nullptr, 0, nullptr, nullptr, 0);
  gemm_generic<<<gconv, 256, 0, stream>>>(wba3, 0, (size_t)kIC * kC, xb, kC, 9, 36, a3_s, a3_b,
                                          concat, 3 * kIC, nullptr, 0, nullptr, nullptr, 0);
  // apr: project concat -> f3 (fused BN+ReLU+*la), also bf16 copy for f4 gemm
  gemm_generic<<<gconv, 256, 0, stream>>>(wbapr, 0, 0, concat, kC5, 1, 0, apr_s, apr_b,
                                          nullptr, 0, dout, 0, f3b, labuf, 1);

  // ---- global grained: f4 = f3 x gan ----
  gan_kernel<<<4096, 256, 0, stream>>>(Abuf, csum2, ganb);
  gemm_generic<<<gconv, 256, 0, stream>>>(f3b, (size_t)kIC * kHW, 0, ganb, kHW, 1, 0,
                                          nullptr, nullptr, nullptr, 0, dout, kIC,
                                          nullptr, nullptr, 2);
}
